// MyModel_21114059227296
// MI455X (gfx1250) — compile-verified, matrix operand order checked by reading, not run on hardware
//
#include <hip/hip_runtime.h>
#include <hip/hip_bf16.h>

#define H_DIM 1024
#define I_DIM 704
#define E_NUM 64
#define TOPK 8
#define NGRP 8
#define TOPKG 4
#define T_NUM 1024
#define SI_DIM 1408
#define RSCALE 2.5f

// padded LDS row strides (bf16 elements): pad +8 keeps ds_load_b128 bank-conflict free
#define STR_H  (H_DIM + 8)     // 1032 -> 2064B -> 516 dwords, 516%64==4
#define STR_I  (I_DIM + 8)     // 712  -> 1424B -> 356 dwords, 356%64==36 (coprime*4)
#define STR_S  (SI_DIM + 8)    // 1416 -> 2832B -> 708 dwords, 708%64==4

typedef __attribute__((ext_vector_type(16))) __bf16 v16bf;
typedef __attribute__((ext_vector_type(8)))  float  v8f;

// ---- fragment loaders -------------------------------------------------------
// 16-bit A/B fragment layout (ISA 7.12.2): lane<16 holds K={off..off+7, off+16..off+23}
// with off=0; lanes 16-31 use off=8.  p points at row + kbase + off.
__device__ __forceinline__ v16bf frag_from_f32(const float* __restrict__ p) {
    v16bf r;
#pragma unroll
    for (int i = 0; i < 8; ++i) {
        r[i]     = (__bf16)p[i];
        r[i + 8] = (__bf16)p[i + 16];
    }
    return r;
}
__device__ __forceinline__ v16bf frag_from_bf16(const __bf16* __restrict__ p) {
    v16bf r;
#pragma unroll
    for (int i = 0; i < 8; ++i) {
        r[i]     = p[i];
        r[i + 8] = p[i + 16];
    }
    return r;
}

__device__ __forceinline__ float fast_sigmoid(float v) {
    return __builtin_amdgcn_rcpf(1.0f + __expf(-v));   // v_rcp_f32, no IEEE div chain
}

// ---- 1. routing -------------------------------------------------------------
__global__ __launch_bounds__(E_NUM)
void route_kernel(const float* __restrict__ x, const float* __restrict__ gw,
                  const float* __restrict__ gbias,
                  int* __restrict__ topk_idx, float* __restrict__ topk_w,
                  int* __restrict__ counts) {
    const int t = blockIdx.x;
    const int e = threadIdx.x;                 // 0..63, one expert per thread
    const float* xp = x + (size_t)t * H_DIM;
    const float* wp = gw + (size_t)e * H_DIM;
    float acc = 0.f;
    for (int k = 0; k < H_DIM; k += 4) {
        acc += xp[k] * wp[k] + xp[k+1] * wp[k+1] + xp[k+2] * wp[k+2] + xp[k+3] * wp[k+3];
    }
    const float score = fast_sigmoid(acc);

    __shared__ float s_score[E_NUM];
    __shared__ float s_sc[E_NUM];
    __shared__ float s_grp[NGRP];
    __shared__ int   s_gsel[NGRP];
    s_score[e] = score;
    s_sc[e]    = score + gbias[e];
    __syncthreads();

    if (e < NGRP) {                            // per-group top-2 sum
        float m1 = -1e30f, m2 = -1e30f;
        for (int j = 0; j < E_NUM / NGRP; ++j) {
            float v = s_sc[e * (E_NUM / NGRP) + j];
            if (v > m1) { m2 = m1; m1 = v; } else if (v > m2) { m2 = v; }
        }
        s_grp[e] = m1 + m2;
        s_gsel[e] = 0;
    }
    __syncthreads();
    if (e == 0) {                              // top-4 groups
        for (int r = 0; r < TOPKG; ++r) {
            int best = 0; float bv = -1e30f;
            for (int g = 0; g < NGRP; ++g)
                if (!s_gsel[g] && s_grp[g] > bv) { bv = s_grp[g]; best = g; }
            s_gsel[best] = 1;
        }
    }
    __syncthreads();
    s_sc[e] = s_gsel[e >> 3] ? s_sc[e] : -1e30f;   // mask unselected groups
    __syncthreads();
    if (e == 0) {                              // top-8 experts + normalize
        int   idx[TOPK];
        float w[TOPK];
        float sum = 0.f;
        for (int r = 0; r < TOPK; ++r) {
            int best = 0; float bv = -1e31f;
            for (int j = 0; j < E_NUM; ++j) {
                float v = s_sc[j];
                if (v > bv) { bv = v; best = j; }
            }
            s_sc[best] = -1e32f;
            idx[r] = best;
            w[r]   = s_score[best];             // gather from UNbiased scores
            sum   += w[r];
        }
        const float inv = RSCALE / (sum + 1e-20f);
        for (int r = 0; r < TOPK; ++r) {
            topk_idx[t * TOPK + r] = idx[r];
            topk_w[t * TOPK + r]   = w[r] * inv;
            atomicAdd(&counts[idx[r]], 1);
        }
    }
}

// ---- 2. prefix sum ----------------------------------------------------------
__global__ void prefix_kernel(const int* __restrict__ counts,
                              int* __restrict__ offsets, int* __restrict__ cursor) {
    if (threadIdx.x == 0 && blockIdx.x == 0) {
        int s = 0;
        for (int e = 0; e < E_NUM; ++e) { offsets[e] = s; s += counts[e]; cursor[e] = 0; }
    }
}

// ---- 3. scatter token slots -------------------------------------------------
__global__ __launch_bounds__(256)
void scatter_kernel(const int* __restrict__ topk_idx, const float* __restrict__ topk_w,
                    const int* __restrict__ offsets, int* __restrict__ cursor,
                    int* __restrict__ slot_token, float* __restrict__ slot_w) {
    const int i = blockIdx.x * blockDim.x + threadIdx.x;   // (t, k) pair
    if (i >= T_NUM * TOPK) return;
    const int t = i / TOPK;
    const int e = topk_idx[i];
    const int pos  = atomicAdd(&cursor[e], 1);
    const int slot = offsets[e] + pos;
    slot_token[slot] = t;
    slot_w[slot]     = topk_w[i];
}

// ---- 4. routed gate+up ------------------------------------------------------
// block = (expert e, 64-col slice of I). Stage Wg/Wu slice in LDS as bf16 ONCE,
// then loop all token m-tiles of this expert. 8 waves: n-slice = wave&3 (4 x 16
// cols), m-tiles start wave>>2 step 2.
__global__ __launch_bounds__(256)
void moe_gate_up(const float* __restrict__ x, const float* __restrict__ Wg,
                 const float* __restrict__ Wu, const int* __restrict__ slot_token,
                 const int* __restrict__ offsets, const int* __restrict__ counts,
                 __bf16* __restrict__ act) {
    const int e   = blockIdx.x;
    const int n_e = counts[e];
    if (n_e == 0) return;
    const int n0  = blockIdx.y * 64;

    extern __shared__ __bf16 smem[];
    __bf16* sWg = smem;
    __bf16* sWu = smem + 64 * STR_H;

    // ---- stage weights: 64 rows x H floats each, fp32 -> bf16 ----
    const int tid = threadIdx.x;
    for (int idx = tid * 4; idx < 64 * H_DIM; idx += 256 * 4) {
        const int r = idx >> 10;          // / H_DIM
        const int k = idx & (H_DIM - 1);
        const size_t gbase = ((size_t)e * I_DIM + n0 + r) * H_DIM + k;
        const float4 g4 = *(const float4*)(Wg + gbase);
        const float4 u4 = *(const float4*)(Wu + gbase);
        const int so = r * STR_H + k;
        sWg[so + 0] = (__bf16)g4.x; sWg[so + 1] = (__bf16)g4.y;
        sWg[so + 2] = (__bf16)g4.z; sWg[so + 3] = (__bf16)g4.w;
        sWu[so + 0] = (__bf16)u4.x; sWu[so + 1] = (__bf16)u4.y;
        sWu[so + 2] = (__bf16)u4.z; sWu[so + 3] = (__bf16)u4.w;
    }
    __syncthreads();

    const int wave = tid >> 5;
    const int lane = tid & 31;
    const int half = lane >> 4;
    const int l16  = lane & 15;
    const int off  = half * 8;
    const int ns   = wave & 3;                 // 16-col slice within 64
    const int nb   = n0 + ns * 16;             // global col base
    const int offs_e = offsets[e];

    const __bf16* bgp = sWg + (size_t)(ns * 16 + l16) * STR_H + off;
    const __bf16* bup = sWu + (size_t)(ns * 16 + l16) * STR_H + off;

    for (int mt = (wave >> 2); mt * 16 < n_e; mt += 2) {
        const int rloc = mt * 16 + l16;
        const int rcl  = rloc < n_e ? rloc : n_e - 1;
        const int tok  = slot_token[offs_e + rcl];
        const float* ap = x + (size_t)tok * H_DIM + off;

        v8f cg = {}, cu = {};
        for (int kb = 0; kb < H_DIM; kb += 32) {
            __builtin_prefetch(ap + kb + 64, 0, 1);
            v16bf a  = frag_from_f32(ap + kb);
            v16bf bg = frag_from_bf16(bgp + kb);
            v16bf bu = frag_from_bf16(bup + kb);
            cg = __builtin_amdgcn_wmma_f32_16x16x32_bf16(false, a, false, bg, (short)0, cg, false, false);
            cu = __builtin_amdgcn_wmma_f32_16x16x32_bf16(false, a, false, bu, (short)0, cu, false, false);
        }
        const int slotbase = offs_e + mt * 16;
#pragma unroll
        for (int r = 0; r < 8; ++r) {
            const int mr = r + 8 * half;       // D layout: col=l16, row=r+8*half
            if (mt * 16 + mr < n_e) {
                const float g = cg[r];
                act[(size_t)(slotbase + mr) * I_DIM + nb + l16] =
                    (__bf16)(g * fast_sigmoid(g) * cu[r]);
            }
        }
    }
}

// ---- 5. routed down + weighted combine --------------------------------------
// block = (expert e, 128-col slice of H). Stage Wd slice bf16 in LDS, loop m-tiles.
// 8 waves: h-slice = wave (8 x 16 cols).
__global__ __launch_bounds__(256)
void moe_down(const __bf16* __restrict__ act, const float* __restrict__ Wd,
              const int* __restrict__ slot_token, const float* __restrict__ slot_w,
              const int* __restrict__ offsets, const int* __restrict__ counts,
              float* __restrict__ out) {
    const int e   = blockIdx.x;
    const int n_e = counts[e];
    if (n_e == 0) return;
    const int h0  = blockIdx.y * 128;

    extern __shared__ __bf16 smem[];
    __bf16* sWd = smem;

    const int tid  = threadIdx.x;
    const int wave = tid >> 5;
    const int lane = tid & 31;

    // ---- stage Wd slice: 128 rows x I floats, fp32 -> bf16 ----
    for (int r = wave; r < 128; r += 8) {
        const size_t gbase = ((size_t)e * H_DIM + h0 + r) * I_DIM;
        for (int k = lane * 4; k < I_DIM; k += 128) {
            const float4 w4 = *(const float4*)(Wd + gbase + k);
            const int so = r * STR_I + k;
            sWd[so + 0] = (__bf16)w4.x; sWd[so + 1] = (__bf16)w4.y;
            sWd[so + 2] = (__bf16)w4.z; sWd[so + 3] = (__bf16)w4.w;
        }
    }
    __syncthreads();

    const int half = lane >> 4;
    const int l16  = lane & 15;
    const int off  = half * 8;
    const int offs_e = offsets[e];
    const int hb = h0 + wave * 16;

    const __bf16* bp = sWd + (size_t)(wave * 16 + l16) * STR_I + off;

    for (int mt = 0; mt * 16 < n_e; ++mt) {
        const int slotbase = offs_e + mt * 16;
        const int rcl = (mt * 16 + l16 < n_e) ? l16 : (n_e - 1 - mt * 16);
        const __bf16* ap = act + (size_t)(slotbase + rcl) * I_DIM + off;

        v8f c = {};
        for (int kb = 0; kb < I_DIM; kb += 32) {       // 22 K-steps
            v16bf a = frag_from_bf16(ap + kb);
            v16bf b = frag_from_bf16(bp + kb);
            c = __builtin_amdgcn_wmma_f32_16x16x32_bf16(false, a, false, b, (short)0, c, false, false);
        }
#pragma unroll
        for (int r = 0; r < 8; ++r) {
            const int mr = r + 8 * half;
            if (mt * 16 + mr < n_e) {
                const int slot = slotbase + mr;
                const int tok  = slot_token[slot];
                atomicAdd(&out[(size_t)tok * H_DIM + hb + l16], c[r] * slot_w[slot]);
            }
        }
    }
}

// ---- 6. shared expert gate+up ----------------------------------------------
// block = 64-col slice of SI. Stage Sg/Su slice bf16, loop all 64 token tiles.
__global__ __launch_bounds__(256)
void shared_gate_up(const float* __restrict__ x, const float* __restrict__ Sg,
                    const float* __restrict__ Su, __bf16* __restrict__ act_s) {
    const int n0 = blockIdx.x * 64;

    extern __shared__ __bf16 smem[];
    __bf16* sWg = smem;
    __bf16* sWu = smem + 64 * STR_H;

    const int tid = threadIdx.x;
    for (int idx = tid * 4; idx < 64 * H_DIM; idx += 256 * 4) {
        const int r = idx >> 10;
        const int k = idx & (H_DIM - 1);
        const size_t gbase = ((size_t)(n0 + r)) * H_DIM + k;
        const float4 g4 = *(const float4*)(Sg + gbase);
        const float4 u4 = *(const float4*)(Su + gbase);
        const int so = r * STR_H + k;
        sWg[so + 0] = (__bf16)g4.x; sWg[so + 1] = (__bf16)g4.y;
        sWg[so + 2] = (__bf16)g4.z; sWg[so + 3] = (__bf16)g4.w;
        sWu[so + 0] = (__bf16)u4.x; sWu[so + 1] = (__bf16)u4.y;
        sWu[so + 2] = (__bf16)u4.z; sWu[so + 3] = (__bf16)u4.w;
    }
    __syncthreads();

    const int wave = tid >> 5;
    const int lane = tid & 31;
    const int half = lane >> 4;
    const int l16  = lane & 15;
    const int off  = half * 8;
    const int ns   = wave & 3;
    const int nb   = n0 + ns * 16;

    const __bf16* bgp = sWg + (size_t)(ns * 16 + l16) * STR_H + off;
    const __bf16* bup = sWu + (size_t)(ns * 16 + l16) * STR_H + off;

    for (int mt = (wave >> 2); mt < T_NUM / 16; mt += 2) {
        const int tok = mt * 16 + l16;
        const float* ap = x + (size_t)tok * H_DIM + off;

        v8f cg = {}, cu = {};
        for (int kb = 0; kb < H_DIM; kb += 32) {
            __builtin_prefetch(ap + kb + 64, 0, 1);
            v16bf a  = frag_from_f32(ap + kb);
            v16bf bg = frag_from_bf16(bgp + kb);
            v16bf bu = frag_from_bf16(bup + kb);
            cg = __builtin_amdgcn_wmma_f32_16x16x32_bf16(false, a, false, bg, (short)0, cg, false, false);
            cu = __builtin_amdgcn_wmma_f32_16x16x32_bf16(false, a, false, bu, (short)0, cu, false, false);
        }
#pragma unroll
        for (int r = 0; r < 8; ++r) {
            const int mr = r + 8 * half;
            const float g = cg[r];
            act_s[(size_t)(mt * 16 + mr) * SI_DIM + nb + l16] =
                (__bf16)(g * fast_sigmoid(g) * cu[r]);
        }
    }
}

// ---- 7. shared expert down --------------------------------------------------
// block = 64-col slice of H. Stage Sd slice bf16 (64 x SI), loop token tiles.
__global__ __launch_bounds__(256)
void shared_down(const __bf16* __restrict__ act_s, const float* __restrict__ Sd,
                 float* __restrict__ out) {
    const int h0 = blockIdx.x * 64;

    extern __shared__ __bf16 smem[];
    __bf16* sWd = smem;

    const int tid  = threadIdx.x;
    const int wave = tid >> 5;
    const int lane = tid & 31;

    for (int r = wave; r < 64; r += 8) {
        const size_t gbase = ((size_t)(h0 + r)) * SI_DIM;
        for (int k = lane * 4; k < SI_DIM; k += 128) {   // 11 exact passes
            const float4 w4 = *(const float4*)(Sd + gbase + k);
            const int so = r * STR_S + k;
            sWd[so + 0] = (__bf16)w4.x; sWd[so + 1] = (__bf16)w4.y;
            sWd[so + 2] = (__bf16)w4.z; sWd[so + 3] = (__bf16)w4.w;
        }
    }
    __syncthreads();

    const int half = lane >> 4;
    const int l16  = lane & 15;
    const int off  = half * 8;
    const int ns   = wave & 3;                 // 4 slices of 16 cols
    const int hb   = h0 + ns * 16;

    const __bf16* bp = sWd + (size_t)(ns * 16 + l16) * STR_S + off;

    for (int mt = (wave >> 2); mt < T_NUM / 16; mt += 2) {
        const __bf16* ap = act_s + (size_t)(mt * 16 + l16) * SI_DIM + off;

        v8f c = {};
        for (int kb = 0; kb < SI_DIM; kb += 32) {       // 44 K-steps
            v16bf a = frag_from_bf16(ap + kb);
            v16bf b = frag_from_bf16(bp + kb);
            c = __builtin_amdgcn_wmma_f32_16x16x32_bf16(false, a, false, b, (short)0, c, false, false);
        }
#pragma unroll
        for (int r = 0; r < 8; ++r) {
            const int mr = r + 8 * half;
            atomicAdd(&out[(size_t)(mt * 16 + mr) * H_DIM + hb + l16], c[r]);
        }
    }
}

// ---- host launch ------------------------------------------------------------
static inline size_t align256(size_t x) { return (x + 255) & ~(size_t)255; }

extern "C" void kernel_launch(void* const* d_in, const int* in_sizes, int n_in,
                              void* d_out, int out_size, void* d_ws, size_t ws_size,
                              hipStream_t stream) {
    const float* x     = (const float*)d_in[0];   // [T, H]
    const float* gw    = (const float*)d_in[1];   // [E, H]
    const float* gbias = (const float*)d_in[2];   // [E]
    const float* Wg    = (const float*)d_in[3];   // [E, I, H]
    const float* Wu    = (const float*)d_in[4];   // [E, I, H]
    const float* Wd    = (const float*)d_in[5];   // [E, H, I]
    const float* Sg    = (const float*)d_in[6];   // [SI, H]
    const float* Su    = (const float*)d_in[7];   // [SI, H]
    const float* Sd    = (const float*)d_in[8];   // [H, SI]
    float* out = (float*)d_out;

    // workspace carve-up
    char* w = (char*)d_ws;
    size_t o = 0;
    int*   counts     = (int*)(w + o); o = align256(o + E_NUM * sizeof(int));
    int*   cursor     = (int*)(w + o); o = align256(o + E_NUM * sizeof(int));
    int*   offsets    = (int*)(w + o); o = align256(o + E_NUM * sizeof(int));
    int*   topk_idx   = (int*)(w + o); o = align256(o + (size_t)T_NUM * TOPK * sizeof(int));
    float* topk_w     = (float*)(w + o); o = align256(o + (size_t)T_NUM * TOPK * sizeof(float));
    int*   slot_token = (int*)(w + o); o = align256(o + (size_t)T_NUM * TOPK * sizeof(int));
    float* slot_w     = (float*)(w + o); o = align256(o + (size_t)T_NUM * TOPK * sizeof(float));
    __bf16* act       = (__bf16*)(w + o); o = align256(o + (size_t)T_NUM * TOPK * I_DIM * sizeof(__bf16));
    __bf16* act_s     = (__bf16*)(w + o); o = align256(o + (size_t)T_NUM * SI_DIM * sizeof(__bf16));
    (void)o; (void)ws_size; (void)in_sizes; (void)n_in;

    hipMemsetAsync(out, 0, (size_t)out_size * sizeof(float), stream);
    hipMemsetAsync(counts, 0, E_NUM * sizeof(int), stream);

    // routing
    route_kernel<<<dim3(T_NUM), dim3(E_NUM), 0, stream>>>(x, gw, gbias, topk_idx, topk_w, counts);
    prefix_kernel<<<dim3(1), dim3(64), 0, stream>>>(counts, offsets, cursor);
    scatter_kernel<<<dim3((T_NUM * TOPK + 255) / 256), dim3(256), 0, stream>>>(
        topk_idx, topk_w, offsets, cursor, slot_token, slot_w);

    // dynamic LDS sizes (bytes)
    const size_t GU_LDS  = (size_t)2 * 64 * STR_H * sizeof(__bf16);   // ~258 KB
    const size_t DN_LDS  = (size_t)128 * STR_I * sizeof(__bf16);      // ~178 KB
    const size_t SGU_LDS = GU_LDS;                                    // ~258 KB
    const size_t SDN_LDS = (size_t)64 * STR_S * sizeof(__bf16);       // ~177 KB

    // routed experts: gate+up then down  (weights staged once per block)
    moe_gate_up<<<dim3(E_NUM, I_DIM / 64), dim3(256), GU_LDS, stream>>>(
        x, Wg, Wu, slot_token, offsets, counts, act);
    moe_down<<<dim3(E_NUM, H_DIM / 128), dim3(256), DN_LDS, stream>>>(
        act, Wd, slot_token, slot_w, offsets, counts, out);

    // shared expert
    shared_gate_up<<<dim3(SI_DIM / 64), dim3(256), SGU_LDS, stream>>>(x, Sg, Su, act_s);
    shared_down<<<dim3(H_DIM / 64), dim3(256), SDN_LDS, stream>>>(act_s, Sd, out);
}
